// BasicMoEBlock_58926951301263
// MI455X (gfx1250) — matrix operand order repaired and checked by reading, not compiled, not run on hardware
//
#include <hip/hip_runtime.h>
#include <hip/hip_bf16.h>

typedef _Float16 half_t;
typedef __attribute__((ext_vector_type(16))) _Float16 v16h;
typedef __attribute__((ext_vector_type(8)))  _Float16 v8h;
typedef __attribute__((ext_vector_type(8)))  float    v8f;
typedef __attribute__((ext_vector_type(4)))  int      v4i;

#define BATCH 32
#define CH    256      // Cin == Cout
#define HDIM  32
#define HW    1024
#define KDIM  2304     // Cin * 9
#define NEXP  4
#define WSZ   589824   // Cout * Cin * 9  (per-expert flat size)

// CDNA5 async global->LDS path (ASYNCcnt); fall back to register-staged copies.
#if defined(__has_builtin)
#if __has_builtin(__builtin_amdgcn_global_load_async_to_lds_b128)
#define ASYNC_LDS 1
#endif
#endif

#if defined(ASYNC_LDS)
__device__ __forceinline__ void async_b128(const half_t* g, half_t* l) {
    __builtin_amdgcn_global_load_async_to_lds_b128(
        (__attribute__((address_space(1))) v4i*)(g),
        (__attribute__((address_space(3))) v4i*)(l), 0, 0);
}
__device__ __forceinline__ void wait_async() {
#if __has_builtin(__builtin_amdgcn_s_wait_asynccnt)
    __builtin_amdgcn_s_wait_asynccnt(0);
#else
    asm volatile("s_wait_asynccnt 0x0" ::: "memory");
#endif
}
#endif

// ---------------------------------------------------------------- cast f32->f16
__global__ void cast_f32_to_f16(const float* __restrict__ in,
                                half_t* __restrict__ out, int n) {
    int i = blockIdx.x * blockDim.x + threadIdx.x;
    int stride = gridDim.x * blockDim.x;
    for (; i < n; i += stride) out[i] = (half_t)in[i];
}

// ---------------------------------------------------------------- channel mean
template <typename T>
__global__ void pool_mean(const T* __restrict__ in, float* __restrict__ pooled) {
    int b = blockIdx.x;
    int c = threadIdx.x;
    const T* p = in + ((size_t)b * CH + c) * HW;
    float s = 0.f;
    for (int i = 0; i < HW; ++i) s += (float)p[i];
    pooled[b * CH + c] = s * (1.0f / HW);
}

// ---------------------------------------------------------------- routing
// rt = sigmoid(pooled @ rW^T + rb) -> group-avg(64) -> rw[b][4]
__global__ void routing_kernel(const float* __restrict__ pooled,
                               const float* __restrict__ rW,
                               const float* __restrict__ rb,
                               float* __restrict__ rw) {
    __shared__ float sig[256];
    int b = blockIdx.x;
    int j = threadIdx.x;
    const float* pv = pooled + b * CH;
    const float* wr = rW + (size_t)j * CH;
    float acc = rb[j];
    for (int c = 0; c < CH; ++c) acc += pv[c] * wr[c];
    sig[j] = 1.0f / (1.0f + __expf(-acc));
    __syncthreads();
    if (j < NEXP) {
        float s = 0.f;
        for (int t = 0; t < 64; ++t) s += sig[j * 64 + t];
        rw[b * NEXP + j] = s * (1.0f / 64.0f);
    }
}

// ---------------------------------------------------------------- expert combine
// Wh[b][j] = f16( sum_e rw[b][e] * ew[e][j] ); layout j = co*KDIM + k (== GEMM W[n][k])
__global__ void combine_weights(const float* __restrict__ rw,
                                const float* __restrict__ ew,
                                half_t* __restrict__ wh) {
    size_t idx    = (size_t)blockIdx.x * blockDim.x + threadIdx.x;
    size_t total  = (size_t)BATCH * WSZ;
    size_t stride = (size_t)gridDim.x * blockDim.x;
    for (; idx < total; idx += stride) {
        int    b = (int)(idx / WSZ);
        size_t j = idx - (size_t)b * WSZ;
        float r0 = rw[b * 4 + 0], r1 = rw[b * 4 + 1];
        float r2 = rw[b * 4 + 2], r3 = rw[b * 4 + 3];
        float v = r0 * ew[j] + r1 * ew[(size_t)WSZ + j] +
                  r2 * ew[2 * (size_t)WSZ + j] + r3 * ew[3 * (size_t)WSZ + j];
        wh[idx] = (half_t)v;
    }
}

// ---------------------------------------------------------------- implicit-GEMM conv
// out[m][n] = sum_k A_im2col[m][k] * W[n][k]   (m = h*32+w, n = co, k = ci*9+kh*3+kw)
// Block: 256 thr = 8 waves; tile M=64, N=256; each wave 32x64 = 2x4 WMMA tiles.
// Double-buffered LDS; A gathered (prefetched to regs), B staged via async->LDS.
template <bool FINAL>
__global__ __launch_bounds__(256)
void cond_conv_wmma(const half_t* __restrict__ A,     // [B][CH][HW] f16
                    const half_t* __restrict__ Wh,    // [B][CH][KDIM] f16
                    const float* __restrict__ gamma, const float* __restrict__ beta,
                    const float* __restrict__ mean,  const float* __restrict__ var,
                    const float* __restrict__ residual, // x (FINAL only)
                    half_t* __restrict__ out_h,         // mid  (!FINAL)
                    float*  __restrict__ out_f)         // final (FINAL)
{
    __shared__ half_t ldsA[2][64 * 40];    // 64 x 32 tiles, rows padded to 40 halves
    __shared__ half_t ldsB[2][256 * 40];   // 256 x 32 tiles, rows padded to 40 halves

    const int b     = blockIdx.y;
    const int mbase = blockIdx.x * 64;
    const int tid   = threadIdx.x;
    const int lane  = tid & 31;
    const int wave  = tid >> 5;
    const int wm    = wave & 1;                // M wave 0..1
    const int wn    = wave >> 1;               // N wave 0..3
    const int rowm  = lane & 15;
    const int kbA   = (lane >> 4) << 3;        // A frag: 0 or 8  (K 0-7,16-23 / 8-15,24-31)
    const int kbB   = (lane >> 4) << 3;        // B frag: 0 or 8  (K 0-7,16-23 / 8-15,24-31)

    v8f acc[2][4] = {};

    // loader roles
    const int a_m  = tid & 63;                 // local m row
    const int a_kg = tid >> 6;                 // 0..3 -> 8 k-values each
    const int gm   = mbase + a_m;
    const int gh   = gm >> 5;
    const int gw   = gm & 31;
    const half_t* Abase = A + (size_t)b * CH * HW;
    const half_t* Wrow  = Wh + ((size_t)b * CH + tid) * KDIM;   // n == tid

    const int NIT = KDIM / 32;                 // 72 K-steps

    // ---- preload tile 0
    {
        #pragma unroll
        for (int j = 0; j < 8; ++j) {
            int k  = a_kg * 8 + j;
            int ci = k / 9;
            int r  = k - ci * 9;
            int kh = r / 3;
            int kw = r - kh * 3;
            int ih = gh + kh - 1;
            int iw = gw + kw - 1;
            half_t v = (half_t)0.f;
            if ((unsigned)ih < 32u && (unsigned)iw < 32u)
                v = Abase[((size_t)ci << 10) + (ih << 5) + iw];
            ldsA[0][a_m * 40 + a_kg * 8 + j] = v;
        }
        const int4* src = (const int4*)(Wrow);
        int4* dst = (int4*)(&ldsB[0][tid * 40]);
        dst[0] = src[0]; dst[1] = src[1]; dst[2] = src[2]; dst[3] = src[3];
    }
    __syncthreads();

    for (int it = 0; it < NIT; ++it) {
        const int cur = it & 1;
        const int nxt = cur ^ 1;
        const int kn  = (it + 1) * 32;
        const bool have_next = (it + 1 < NIT);

        // ---- issue next-tile loads early (hide latency behind WMMA)
        half_t areg[8];
#if !defined(ASYNC_LDS)
        int4 breg[4];
#endif
        if (have_next) {
            #pragma unroll
            for (int j = 0; j < 8; ++j) {
                int k  = kn + a_kg * 8 + j;
                int ci = k / 9;
                int r  = k - ci * 9;
                int kh = r / 3;
                int kw = r - kh * 3;
                int ih = gh + kh - 1;
                int iw = gw + kw - 1;
                half_t v = (half_t)0.f;
                if ((unsigned)ih < 32u && (unsigned)iw < 32u)
                    v = Abase[((size_t)ci << 10) + (ih << 5) + iw];
                areg[j] = v;
            }
#if defined(ASYNC_LDS)
            #pragma unroll
            for (int i = 0; i < 4; ++i)
                async_b128(Wrow + kn + i * 8, &ldsB[nxt][tid * 40 + i * 8]);
#else
            {
                const int4* src = (const int4*)(Wrow + kn);
                breg[0] = src[0]; breg[1] = src[1];
                breg[2] = src[2]; breg[3] = src[3];
            }
#endif
        }

        // ---- fragments from current tile + WMMA
        v16h afrag[2], bfrag[4];
        #pragma unroll
        for (int i = 0; i < 2; ++i) {
            int ml = wm * 32 + i * 16 + rowm;
            v8h lo = *(const v8h*)(&ldsA[cur][ml * 40 + kbA]);
            v8h hi = *(const v8h*)(&ldsA[cur][ml * 40 + kbA + 16]);
            afrag[i] = __builtin_shufflevector(lo, hi, 0,1,2,3,4,5,6,7,8,9,10,11,12,13,14,15);
        }
        #pragma unroll
        for (int j = 0; j < 4; ++j) {
            int nl = wn * 64 + j * 16 + rowm;
            v8h lo = *(const v8h*)(&ldsB[cur][nl * 40 + kbB]);
            v8h hi = *(const v8h*)(&ldsB[cur][nl * 40 + kbB + 16]);
            bfrag[j] = __builtin_shufflevector(lo, hi, 0,1,2,3,4,5,6,7,8,9,10,11,12,13,14,15);
        }
        #pragma unroll
        for (int i = 0; i < 2; ++i)
            #pragma unroll
            for (int j = 0; j < 4; ++j)
                acc[i][j] = __builtin_amdgcn_wmma_f32_16x16x32_f16(
                    false, afrag[i], false, bfrag[j],
                    (short)0, acc[i][j], false, false);

        // ---- commit next tile to LDS
        if (have_next) {
            #pragma unroll
            for (int j = 0; j < 8; ++j)
                ldsA[nxt][a_m * 40 + a_kg * 8 + j] = areg[j];
#if !defined(ASYNC_LDS)
            int4* dst = (int4*)(&ldsB[nxt][tid * 40]);
            dst[0] = breg[0]; dst[1] = breg[1];
            dst[2] = breg[2]; dst[3] = breg[3];
#endif
        }
#if defined(ASYNC_LDS)
        wait_async();
#endif
        __syncthreads();
    }

    // ---- epilogue: BN fold (+ReLU, +residual for final)
    #pragma unroll
    for (int j = 0; j < 4; ++j) {
        int n = wn * 64 + j * 16 + rowm;                 // co; C/D: N = lane%16
        float inv  = gamma[n] * rsqrtf(var[n] + 1e-5f);
        float bias = beta[n] - mean[n] * inv;
        #pragma unroll
        for (int i = 0; i < 2; ++i) {
            int m0 = mbase + wm * 32 + i * 16 + ((lane >> 4) << 3); // 8 consecutive m
            size_t base = ((size_t)b * CH + n) * HW + m0;
            if (FINAL) {
                #pragma unroll
                for (int v = 0; v < 8; ++v) {
                    float val = acc[i][j][v] * inv + bias + residual[base + v];
                    out_f[base + v] = val > 0.f ? val : 0.f;
                }
            } else {
                v8h hv;
                #pragma unroll
                for (int v = 0; v < 8; ++v) {
                    float val = acc[i][j][v] * inv + bias;
                    val = val > 0.f ? val : 0.f;
                    hv[v] = (half_t)val;
                }
                *(v8h*)(&out_h[base]) = hv;
            }
        }
    }
}

// ---------------------------------------------------------------- launch
extern "C" void kernel_launch(void* const* d_in, const int* in_sizes, int n_in,
                              void* d_out, int out_size, void* d_ws, size_t ws_size,
                              hipStream_t stream) {
    const float* x         = (const float*)d_in[0];
    const float* r1_W      = (const float*)d_in[1];
    const float* r1_b      = (const float*)d_in[2];
    const float* e1_w      = (const float*)d_in[3];
    const float* bn1_gamma = (const float*)d_in[4];
    const float* bn1_beta  = (const float*)d_in[5];
    const float* bn1_mean  = (const float*)d_in[6];
    const float* bn1_var   = (const float*)d_in[7];
    const float* r2_W      = (const float*)d_in[8];
    const float* r2_b      = (const float*)d_in[9];
    const float* e2_w      = (const float*)d_in[10];
    const float* bn2_gamma = (const float*)d_in[11];
    const float* bn2_beta  = (const float*)d_in[12];
    const float* bn2_mean  = (const float*)d_in[13];
    const float* bn2_var   = (const float*)d_in[14];

    char* ws = (char*)d_ws;
    const size_t XH_BYTES  = (size_t)BATCH * CH * HW * sizeof(half_t);   // 16 MiB
    const size_t WH_BYTES  = (size_t)BATCH * WSZ * sizeof(half_t);       // 36 MiB
    half_t* xh     = (half_t*)ws;
    half_t* midh   = (half_t*)(ws + XH_BYTES);
    half_t* wh     = (half_t*)(ws + 2 * XH_BYTES);                       // reused for both layers
    float*  pooled = (float*)(ws + 2 * XH_BYTES + WH_BYTES);
    float*  rwv    = (float*)(ws + 2 * XH_BYTES + WH_BYTES + 32768);

    const int nX = BATCH * CH * HW;
    dim3 cg(16, BATCH);   // 16 M-tiles x 32 samples

    // ---- layer 1
    cast_f32_to_f16<<<4096, 256, 0, stream>>>(x, xh, nX);
    pool_mean<float><<<BATCH, 256, 0, stream>>>(x, pooled);
    routing_kernel<<<BATCH, 256, 0, stream>>>(pooled, r1_W, r1_b, rwv);
    combine_weights<<<9216, 256, 0, stream>>>(rwv, e1_w, wh);
    cond_conv_wmma<false><<<cg, 256, 0, stream>>>(
        xh, wh, bn1_gamma, bn1_beta, bn1_mean, bn1_var,
        nullptr, midh, nullptr);

    // ---- layer 2
    pool_mean<half_t><<<BATCH, 256, 0, stream>>>(midh, pooled);
    routing_kernel<<<BATCH, 256, 0, stream>>>(pooled, r2_W, r2_b, rwv);
    combine_weights<<<9216, 256, 0, stream>>>(rwv, e2_w, wh);
    cond_conv_wmma<true><<<cg, 256, 0, stream>>>(
        midh, wh, bn2_gamma, bn2_beta, bn2_mean, bn2_var,
        x, nullptr, (float*)d_out);
}
